// DrugEncoder_58737972740422
// MI455X (gfx1250) — matrix operand order repaired and checked by reading, not compiled, not run on hardware
//
#include <hip/hip_runtime.h>
#include <hip/hip_bf16.h>

typedef __attribute__((ext_vector_type(16))) _Float16 v16h;
typedef __attribute__((ext_vector_type(8)))  _Float16 v8h;
typedef __attribute__((ext_vector_type(8)))  float    v8f;

#define BATCH 256
#define NAT 64
#define NTOK 65           // NA + graph token
#define DMODEL 512
#define NHEAD 16
#define HDIM 32
#define D3 1536
#define MROWS (BATCH * NTOK)   // 16640
#define SQRT_2PI 2.5066262f    // sqrt(2*3.14159)
#define QK_SCALE 0.17677669529663687f  // 1/sqrt(32)

// ---------------------------------------------------------------------------
// Async 16-byte global -> LDS copy (gfx1250 VGLOBAL async op, ASYNCcnt).
// LDS destination is the wave-relative byte offset = low 32 bits of the
// generic pointer to a __shared__ object (addrspacecast(3->flat) keeps the
// LDS offset in the low word).
// ---------------------------------------------------------------------------
__device__ __forceinline__ void async_copy16(const _Float16* g, const _Float16* l) {
  const unsigned lds_off = (unsigned)(unsigned long long)l;
  const unsigned long long ga = (unsigned long long)g;
  asm volatile("global_load_async_to_lds_b128 %0, %1, off"
               :: "v"(lds_off), "v"(ga)
               : "memory");
}

__device__ __forceinline__ void wait_async0() {
  asm volatile("s_wait_asynccnt 0" ::: "memory");
}

// ---------------------------------------------------------------------------
// fp32 -> f16 conversion for weights
// ---------------------------------------------------------------------------
__global__ void f32_to_f16_kernel(const float* __restrict__ src,
                                  _Float16* __restrict__ dst, int n) {
  int i = blockIdx.x * blockDim.x + threadIdx.x;
  if (i < n) dst[i] = (_Float16)src[i];
}

// ---------------------------------------------------------------------------
// Atom embedding: gaussian on continuous channel + 6 table gathers + token row
// One block per (b, n); threads = d. Writes f16 activations [MROWS, DMODEL].
// ---------------------------------------------------------------------------
__global__ __launch_bounds__(512)
void atom_embed_kernel(const int* __restrict__ atom_fea,      // [B,7,64]
                       const float* __restrict__ atom_tables, // [6,100,512]
                       const float* __restrict__ ga_means,    // [512]
                       const float* __restrict__ ga_stds,     // [512]
                       const float* __restrict__ ga_mul,      // scalar
                       const float* __restrict__ ga_bias,     // scalar
                       const float* __restrict__ graph_token, // [512]
                       _Float16* __restrict__ x_h) {          // [MROWS,512]
  const int bn = blockIdx.x;
  const int b = bn / NTOK;
  const int n = bn % NTOK;
  const int d = threadIdx.x;
  const size_t outoff = (size_t)bn * DMODEL + d;
  if (n == 0) {
    x_h[outoff] = (_Float16)graph_token[d];
    return;
  }
  const int a = n - 1;
  __shared__ int fea[7];
  if (d < 7) fea[d] = atom_fea[((size_t)b * 7 + d) * NAT + a];
  __syncthreads();

  float val = 0.0f;
  if (fea[6] != 0) {
    const float cont = (float)fea[6];
    const float g = ga_mul[0] * cont + ga_bias[0];
    const float sd = fabsf(ga_stds[d]) + 1e-5f;
    const float z = (g - ga_means[d]) / sd;
    val = expf(-0.5f * z * z) / (SQRT_2PI * sd);
  }
#pragma unroll
  for (int i = 0; i < 6; ++i)
    val += atom_tables[((size_t)i * 100 + fea[i]) * DMODEL + d];
  x_h[outoff] = (_Float16)val;
}

// ---------------------------------------------------------------------------
// Edge bias: gaussian(dist) + multi-hop path-count table adds + mask + token.
// One block per batch element. Bits kept as u64 bitmasks in LDS; the 64x64x64
// count matmuls run in VALU against LDS. bias buffer [B, H, 65, 65].
// ---------------------------------------------------------------------------
__global__ __launch_bounds__(256)
void edge_bias_kernel(const int* __restrict__ bond_adj,       // [B,64,64]
                      const float* __restrict__ dist_adj,     // [B,64,64]
                      const float* __restrict__ edge_tables,  // [6,51,16]
                      const float* __restrict__ gb_means,     // [16]
                      const float* __restrict__ gb_stds,      // [16]
                      const float* __restrict__ gb_mul,       // scalar
                      const float* __restrict__ gb_bias,      // scalar
                      const float* __restrict__ edge_graph_token, // [16]
                      float* __restrict__ biasbuf) {          // [B,16,65,65]
  __shared__ int   bond[NAT * NAT];
  __shared__ float jc[NAT * NAT];
  __shared__ float jn[NAT * NAT];
  __shared__ unsigned long long bmask[NAT];

  const int b = blockIdx.x;
  const int tid = threadIdx.x;
  float* bb = biasbuf + (size_t)b * NHEAD * NTOK * NTOK;

  for (int t = tid; t < NAT * NAT; t += 256)
    bond[t] = bond_adj[(size_t)b * NAT * NAT + t];
  __syncthreads();

  // init interior cells: gaussian(dist) + (-inf mask); also token row/col.
  const float gmul = gb_mul[0], gbia = gb_bias[0];
  for (int t = tid; t < NAT * NAT; t += 256) {
    const int y = t >> 6, x = t & 63;
    const float dv = dist_adj[(size_t)b * NAT * NAT + t];
    const float g = gmul * dv + gbia;
    const float mask = (bond[t] != 0) ? 0.0f : -__builtin_inff();
#pragma unroll
    for (int hh = 0; hh < NHEAD; ++hh) {
      float val = 0.0f;
      if (dv != 0.0f) {
        const float sd = fabsf(gb_stds[hh]) + 1e-5f;
        const float z = (g - gb_means[hh]) / sd;
        val = expf(-0.5f * z * z) / (SQRT_2PI * sd);
      }
      bb[((size_t)hh * NTOK + (y + 1)) * NTOK + (x + 1)] = val + mask;
    }
  }
  for (int t = tid; t < NHEAD * NTOK; t += 256) {
    const int hh = t / NTOK, p = t % NTOK;
    const float tv = edge_graph_token[hh];
    bb[((size_t)hh * NTOK + 0) * NTOK + p] = tv;  // token row
    bb[((size_t)hh * NTOK + p) * NTOK + 0] = tv;  // token col
  }
  __syncthreads();

  for (int i = 0; i < 6; ++i) {
    if (tid < NAT) {
      unsigned long long m = 0ULL;
      for (int x = 0; x < NAT; ++x) {
        const int bv = bond[tid * NAT + x];
        const unsigned long long bit =
            (bv > 0) ? (unsigned long long)(((bv - 1) >> i) & 1) : 0ULL;
        m |= bit << x;
      }
      bmask[tid] = m;
    }
    __syncthreads();
    for (int t = tid; t < NAT * NAT; t += 256)
      jc[t] = (float)((bmask[t >> 6] >> (t & 63)) & 1ULL);
    __syncthreads();

    for (int hop = 0; hop < 4; ++hop) {
      if (hop > 0) {
        for (int t = tid; t < NAT * NAT; t += 256) {
          const int y = t >> 6, x = t & 63;
          const float* jr = &jc[y * NAT];
          float s = 0.0f;
          for (int kk = 0; kk < NAT; ++kk)
            s += jr[kk] * (float)((bmask[kk] >> x) & 1ULL);
          jn[t] = s;
        }
        __syncthreads();
        for (int t = tid; t < NAT * NAT; t += 256) jc[t] = jn[t];
        __syncthreads();
      }
      // per-cell table accumulate; each cell owned by a fixed thread -> plain RMW
      for (int t = tid; t < NAT * NAT; t += 256) {
        int id = (int)jc[t];
        id = id < 0 ? 0 : (id > 50 ? 50 : id);
        const float* row = edge_tables + ((size_t)i * 51 + id) * NHEAD;
        const int y = t >> 6, x = t & 63;
#pragma unroll
        for (int hh = 0; hh < NHEAD; ++hh)
          bb[((size_t)hh * NTOK + (y + 1)) * NTOK + (x + 1)] += row[hh];
      }
      __syncthreads();
    }
  }
}

// ---------------------------------------------------------------------------
// WMMA f16 GEMM with async-LDS double buffering:
//   out[M,N] = A[M,K] @ W[N,K]^T + bias
// Block = 128 threads (4 waves) computing a 64x64 C tile. Per 32-wide K chunk
// the block DMAs A[64x32] and W[64x32] into LDS with
// global_load_async_to_lds_b128 (double buffered, drained by s_wait_asynccnt),
// then each wave runs 4 v_wmma_f32_16x16x32_f16 on ds_load_b128 fragments.
// ---------------------------------------------------------------------------
#define KCH 32
#define TILE 64

__global__ __launch_bounds__(128)
void wmma_gemm_kernel(const _Float16* __restrict__ A,   // [M,K]
                      const _Float16* __restrict__ W,   // [N,K]
                      const float* __restrict__ bias,   // [N] or null
                      _Float16* __restrict__ outh,      // f16 out or null
                      float* __restrict__ outf,         // f32 out or null
                      int M, int N, int K) {
  __shared__ alignas(16) _Float16 bufA[2][TILE * KCH];  // 4 KB each
  __shared__ alignas(16) _Float16 bufB[2][TILE * KCH];

  const int tid = threadIdx.x;
  const int wave = tid >> 5;
  const int lane = tid & 31;
  const int hf = lane >> 4;       // half-wave selector
  const int r = lane & 15;
  const int m0 = blockIdx.y * TILE;
  const int n0 = blockIdx.x * TILE;

  const _Float16* Abase = A + (size_t)m0 * K;
  const _Float16* Wbase = W + (size_t)n0 * K;

  // this thread's two (row, seg) DMA slots: 64 rows x 4 segs of 8 halfs
  const int e0 = tid, e1 = tid + 128;
  const int r0 = e0 >> 2, s0 = e0 & 3;
  const int r1 = e1 >> 2, s1 = e1 & 3;

  v8f c0 = {}, c1 = {}, c2 = {}, c3 = {};

  // prologue: stage chunk 0 into buffer 0
  {
    const int k = 0;
    async_copy16(Abase + (size_t)r0 * K + k + s0 * 8, &bufA[0][r0 * KCH + s0 * 8]);
    async_copy16(Wbase + (size_t)r0 * K + k + s0 * 8, &bufB[0][r0 * KCH + s0 * 8]);
    async_copy16(Abase + (size_t)r1 * K + k + s1 * 8, &bufA[0][r1 * KCH + s1 * 8]);
    async_copy16(Wbase + (size_t)r1 * K + k + s1 * 8, &bufB[0][r1 * KCH + s1 * 8]);
  }
  wait_async0();
  __syncthreads();

  const int nch = K / KCH;
  for (int kc = 0; kc < nch; ++kc) {
    const int cur = kc & 1;
    const int nxt = cur ^ 1;

    if (kc + 1 < nch) {
      const int k = (kc + 1) * KCH;
      async_copy16(Abase + (size_t)r0 * K + k + s0 * 8, &bufA[nxt][r0 * KCH + s0 * 8]);
      async_copy16(Wbase + (size_t)r0 * K + k + s0 * 8, &bufB[nxt][r0 * KCH + s0 * 8]);
      async_copy16(Abase + (size_t)r1 * K + k + s1 * 8, &bufA[nxt][r1 * KCH + s1 * 8]);
      async_copy16(Wbase + (size_t)r1 * K + k + s1 * 8, &bufB[nxt][r1 * KCH + s1 * 8]);
      if (kc + 2 < nch)  // warm L2 one chunk further ahead
        __builtin_prefetch(Abase + (size_t)r0 * K + (kc + 2) * KCH + s0 * 8, 0, 1);
    }

    // A fragment for this wave's 16 rows (CDNA5 16-bit A 16x32 layout):
    //  lanes 0-15: K=[k..k+7],[k+16..k+23];  lanes 16-31: +8
    const _Float16* la = &bufA[cur][(wave * 16 + r) * KCH];
    const v8h alo = *(const v8h*)(la + hf * 8);
    const v8h ahi = *(const v8h*)(la + 16 + hf * 8);
    v16h av;
#pragma unroll
    for (int j = 0; j < 8; ++j) { av[j] = alo[j]; av[j + 8] = ahi[j]; }

#pragma unroll
    for (int s = 0; s < 4; ++s) {
      // B fragment (32x16): lanes 0-15 col N=r K=[k..k+7],[k+16..k+23]; lanes 16-31 +8
      const _Float16* lb = &bufB[cur][(s * 16 + r) * KCH];
      const v8h blo = *(const v8h*)(lb + hf * 8);
      const v8h bhi = *(const v8h*)(lb + 16 + hf * 8);
      v16h bv;
#pragma unroll
      for (int j = 0; j < 8; ++j) { bv[j] = blo[j]; bv[j + 8] = bhi[j]; }

      v8f& c = (s == 0) ? c0 : (s == 1) ? c1 : (s == 2) ? c2 : c3;
      c = __builtin_amdgcn_wmma_f32_16x16x32_f16(
          /*neg_a=*/false, av, /*neg_b=*/false, bv,
          /*c_mod=*/(short)0, c, /*reuse_a=*/false, /*reuse_b=*/false);
    }

    wait_async0();      // next chunk resident in LDS
    __syncthreads();    // all waves done reading cur before it is reused
  }

  // D layout: lanes 0-15 -> (M=m0+w*16+j, N), lanes 16-31 -> (M=..+8+j, N)
#pragma unroll
  for (int s = 0; s < 4; ++s) {
    const v8f& c = (s == 0) ? c0 : (s == 1) ? c1 : (s == 2) ? c2 : c3;
    const int col = n0 + s * 16 + r;
    const float bn = bias ? bias[col] : 0.0f;
#pragma unroll
    for (int j = 0; j < 8; ++j) {
      const int row = m0 + wave * 16 + hf * 8 + j;
      const float val = c[j] + bn;
      if (outh) outh[(size_t)row * N + col] = (_Float16)val;
      if (outf) outf[(size_t)row * N + col] = val;
    }
  }
}

// ---------------------------------------------------------------------------
// Attention: one block per (b, head). q/k/v head slices into LDS (q pre-scaled),
// scores + bias, softmax, p@v. Writes f16 context [MROWS, DMODEL].
// ---------------------------------------------------------------------------
__global__ __launch_bounds__(64)
void attention_kernel(const _Float16* __restrict__ qkv_h, // [MROWS, 1536]
                      const float* __restrict__ biasbuf,  // [B,16,65,65]
                      _Float16* __restrict__ o_h) {       // [MROWS, 512]
  __shared__ float qs[NTOK * HDIM];
  __shared__ float ks[NTOK * HDIM];
  __shared__ float vs[NTOK * HDIM];
  __shared__ float sb[NTOK * NTOK];

  const int bh = blockIdx.x;
  const int b = bh >> 4;
  const int h = bh & 15;
  const int tid = threadIdx.x;

  const _Float16* base = qkv_h + (size_t)b * NTOK * D3 + h * HDIM;
  for (int t = tid; t < NTOK * HDIM; t += 64) {
    const int n = t >> 5, d = t & 31;
    const size_t roff = (size_t)n * D3 + d;
    qs[t] = (float)base[roff] * QK_SCALE;
    ks[t] = (float)base[roff + DMODEL];
    vs[t] = (float)base[roff + 2 * DMODEL];
  }
  __syncthreads();

  const float* bbh = biasbuf + (size_t)bh * NTOK * NTOK;
  for (int qi = tid; qi < NTOK; qi += 64) {
    const float* qrow = &qs[qi * HDIM];
    float* srow = &sb[qi * NTOK];
    float mx = -3.4e38f;
    for (int j = 0; j < NTOK; ++j) {
      float s = bbh[(size_t)qi * NTOK + j];
      const float* krow = &ks[j * HDIM];
#pragma unroll
      for (int d = 0; d < HDIM; ++d) s += qrow[d] * krow[d];
      srow[j] = s;
      mx = fmaxf(mx, s);
    }
    float sum = 0.0f;
    for (int j = 0; j < NTOK; ++j) {
      const float e = expf(srow[j] - mx);
      srow[j] = e;
      sum += e;
    }
    const float inv = 1.0f / sum;
    for (int d = 0; d < HDIM; ++d) {
      float o = 0.0f;
      for (int j = 0; j < NTOK; ++j) o += srow[j] * vs[j * HDIM + d];
      o_h[((size_t)b * NTOK + qi) * DMODEL + h * HDIM + d] = (_Float16)(o * inv);
    }
  }
}

// ---------------------------------------------------------------------------
// Host launcher
// ---------------------------------------------------------------------------
extern "C" void kernel_launch(void* const* d_in, const int* in_sizes, int n_in,
                              void* d_out, int out_size, void* d_ws, size_t ws_size,
                              hipStream_t stream) {
  const int*   atom_fea         = (const int*)d_in[0];
  const int*   bond_adj         = (const int*)d_in[1];
  const float* dist_adj         = (const float*)d_in[2];
  const float* atom_tables      = (const float*)d_in[3];
  const float* ga_means         = (const float*)d_in[4];
  const float* ga_stds          = (const float*)d_in[5];
  const float* ga_mul           = (const float*)d_in[6];
  const float* ga_bias          = (const float*)d_in[7];
  const float* graph_token      = (const float*)d_in[8];
  const float* edge_tables      = (const float*)d_in[9];
  const float* gb_means         = (const float*)d_in[10];
  const float* gb_stds          = (const float*)d_in[11];
  const float* gb_mul           = (const float*)d_in[12];
  const float* gb_bias          = (const float*)d_in[13];
  const float* edge_graph_token = (const float*)d_in[14];
  const float* in_proj_w        = (const float*)d_in[15];
  const float* in_proj_b        = (const float*)d_in[16];
  const float* out_proj_w       = (const float*)d_in[17];
  const float* out_proj_b       = (const float*)d_in[18];

  char* ws = (char*)d_ws;
  _Float16* x_h     = (_Float16*)(ws + 0);          // 16640*512*2  = 17,039,360
  _Float16* w1_h    = (_Float16*)(ws + 17039360);   // 1536*512*2   =  1,572,864
  _Float16* w2_h    = (_Float16*)(ws + 18612224);   // 512*512*2    =    524,288
  _Float16* qkv_h   = (_Float16*)(ws + 19136512);   // 16640*1536*2 = 51,118,080
  float*    biasbuf = (float*)   (ws + 70254592);   // 256*16*65*65*4 = 69,222,400
  _Float16* o_h     = (_Float16*)(ws + 139476992);  // 16640*512*2  = 17,039,360
                                                    // total ~156.5 MB

  // 1. weight conversion to f16
  f32_to_f16_kernel<<<(D3 * DMODEL + 255) / 256, 256, 0, stream>>>(
      in_proj_w, w1_h, D3 * DMODEL);
  f32_to_f16_kernel<<<(DMODEL * DMODEL + 255) / 256, 256, 0, stream>>>(
      out_proj_w, w2_h, DMODEL * DMODEL);

  // 2. atom embedding -> x_h
  atom_embed_kernel<<<MROWS, DMODEL, 0, stream>>>(
      atom_fea, atom_tables, ga_means, ga_stds, ga_mul, ga_bias, graph_token, x_h);

  // 3. edge bias -> biasbuf
  edge_bias_kernel<<<BATCH, 256, 0, stream>>>(
      bond_adj, dist_adj, edge_tables, gb_means, gb_stds, gb_mul, gb_bias,
      edge_graph_token, biasbuf);

  // 4. QKV projection (WMMA + async LDS): [16640,512] x [1536,512]^T
  wmma_gemm_kernel<<<dim3(D3 / TILE, MROWS / TILE), 128, 0, stream>>>(
      x_h, w1_h, in_proj_b, qkv_h, nullptr, MROWS, D3, DMODEL);

  // 5. attention per (b, head)
  attention_kernel<<<BATCH * NHEAD, 64, 0, stream>>>(qkv_h, biasbuf, o_h);

  // 6. output projection (WMMA + async LDS) -> d_out fp32
  wmma_gemm_kernel<<<dim3(DMODEL / TILE, MROWS / TILE), 128, 0, stream>>>(
      o_h, w2_h, out_proj_b, nullptr, (float*)d_out, MROWS, DMODEL, DMODEL);
}
